// Block_37374805410060
// MI455X (gfx1250) — matrix operand order repaired and checked by reading, not compiled, not run on hardware
//
#include <hip/hip_runtime.h>

// ---------------------------------------------------------------------------
// Transformer block for MI455X (gfx1250, wave32, WMMA).
// All GEMM FLOPs go through v_wmma_f32_16x16x32_bf16; GEMM A-tiles staged with
// GLOBAL_LOAD_ASYNC_TO_LDS_B128 (ASYNCcnt) when the builtin is available.
// ---------------------------------------------------------------------------

typedef __attribute__((ext_vector_type(16))) __bf16 v16bf;
typedef __attribute__((ext_vector_type(8)))  float  v8f;
typedef int v4i_gcc __attribute__((vector_size(16)));   // matches builtin's V4i

union Frag {            // 16 bf16 values = 8 VGPRs, matches v16bf operand
    v16bf v;
    uint4 q[2];
};

#define NB   2
#define NT   2048
#define NC   768
#define NH   12
#define ND   64
#define NF   3072
#define NTOK (NB * NT)     // 4096
#define LDQKV (3 * NC)     // 2304

#if __has_builtin(__builtin_amdgcn_global_load_async_to_lds_b128)
#define HAVE_ASYNC_LDS 1
#else
#define HAVE_ASYNC_LDS 0
#endif

__device__ __forceinline__ void wait_async0() {
#if __has_builtin(__builtin_amdgcn_s_wait_asynccnt)
    __builtin_amdgcn_s_wait_asynccnt(0);
#else
    asm volatile("s_wait_asynccnt 0" ::: "memory");
#endif
}

#if HAVE_ASYNC_LDS
__device__ __forceinline__ void g2l_async_b128(const unsigned short* g,
                                               unsigned short* l) {
    // generic LDS pointer carries the LDS byte offset in its low 32 bits
    __builtin_amdgcn_global_load_async_to_lds_b128(
        (__attribute__((address_space(1))) v4i_gcc*)(uintptr_t)g,
        (__attribute__((address_space(3))) v4i_gcc*)(unsigned int)(uintptr_t)l,
        0, 0);
}
#endif

__device__ __forceinline__ unsigned short f2bf(float f) {
    unsigned int u = __builtin_bit_cast(unsigned int, f);
    unsigned int r = u + 0x7FFFu + ((u >> 16) & 1u);   // round-to-nearest-even
    return (unsigned short)(r >> 16);
}

__device__ __forceinline__ v8f wmma_bf16(const Frag& a, const Frag& b, v8f c) {
    return __builtin_amdgcn_wmma_f32_16x16x32_bf16(
        false, a.v, false, b.v, (short)0, c, false, false);
}

// ---------------------------------------------------------------------------
// Weight conversion / packing
// ---------------------------------------------------------------------------
__global__ void f2bf_kernel(const float* __restrict__ in,
                            unsigned short* __restrict__ out, int n) {
    int i = blockIdx.x * 256 + threadIdx.x;
    if (i < n) out[i] = f2bf(in[i]);
}

// Wq/Wk/Wv [H,C,D] f32  ->  packed bf16 [C, 3C]  (cols: q heads | k heads | v heads)
__global__ void pack_qkv_kernel(const float* __restrict__ Wq,
                                const float* __restrict__ Wk,
                                const float* __restrict__ Wv,
                                unsigned short* __restrict__ out) {
    int i = blockIdx.x * 256 + threadIdx.x;      // over C * 3C
    if (i >= NC * LDQKV) return;
    int c   = i / LDQKV;
    int j   = i % LDQKV;
    int sel = j / NC;
    int jj  = j % NC;
    int h   = jj >> 6;
    int d   = jj & 63;
    const float* W = (sel == 0) ? Wq : (sel == 1) ? Wk : Wv;
    out[i] = f2bf(W[((size_t)h * NC + c) * ND + d]);
}

// ---------------------------------------------------------------------------
// LayerNorm: one row (C=768) per block of 256, output bf16
// ---------------------------------------------------------------------------
__global__ __launch_bounds__(256)
void ln_kernel(const float* __restrict__ x,
               const float* __restrict__ g,
               const float* __restrict__ bta,
               unsigned short* __restrict__ out) {
    __shared__ float r1[256];
    __shared__ float r2[256];
    const int row = blockIdx.x;
    const int tid = threadIdx.x;
    const float* xr = x + (size_t)row * NC;
    float s = 0.f, s2 = 0.f;
    #pragma unroll
    for (int i = tid; i < NC; i += 256) { float v = xr[i]; s += v; s2 += v * v; }
    r1[tid] = s; r2[tid] = s2;
    __syncthreads();
    for (int off = 128; off > 0; off >>= 1) {
        if (tid < off) { r1[tid] += r1[tid + off]; r2[tid] += r2[tid + off]; }
        __syncthreads();
    }
    const float mu   = r1[0] * (1.f / NC);
    const float var  = r2[0] * (1.f / NC) - mu * mu;
    const float rinv = rsqrtf(var + 1e-5f);
    #pragma unroll
    for (int i = tid; i < NC; i += 256)
        out[(size_t)row * NC + i] = f2bf((xr[i] - mu) * rinv * g[i] + bta[i]);
}

// ---------------------------------------------------------------------------
// Generic bf16 WMMA GEMM:  C[M,N] = A[M,K] * B[K,N]  (+bias, relu, +res)
// Block 256 = 8 waves. Tile M=128 (16 rows/wave), N=64 (4x 16-col subtiles).
// K-step 64 (two WMMA K-chunks per barrier pair -> 8 WMMAs per wave per iter).
// A tile staged with async global->LDS copies (ASYNCcnt) when available.
// ---------------------------------------------------------------------------
template <bool STORE_BF16, bool RELU, bool HAS_BIAS, bool HAS_RES>
__global__ __launch_bounds__(256)
void gemm_bf16(const unsigned short* __restrict__ A,
               const unsigned short* __restrict__ Bm,
               void* __restrict__ Cout,
               const float* __restrict__ bias,
               const float* __restrict__ res,
               int N, int K) {
    __shared__ unsigned short Atile[128 * 64];          // [row][k]       16 KB
    __shared__ unsigned short Btile[2 * 4 * 16 * 32];   // [kc][nt][col][k] 8 KB
    const int tid  = threadIdx.x;
    const int wave = tid >> 5;
    const int lane = tid & 31;
    const int hlf  = lane >> 4;
    const int l15  = lane & 15;
    const int nBase = blockIdx.x * 64;
    const int mBase = blockIdx.y * 128;

    v8f acc[4];
    #pragma unroll
    for (int nt = 0; nt < 4; ++nt)
        #pragma unroll
        for (int r = 0; r < 8; ++r) acc[nt][r] = 0.f;

    for (int k0 = 0; k0 < K; k0 += 64) {
        __syncthreads();
        {   // A tile: 128x64 bf16, 2 threads per row (32 ushorts each)
            const int row = tid >> 1;
            const int seg = (tid & 1) * 32;
            const unsigned short* src = A + (size_t)(mBase + row) * K + k0 + seg;
            unsigned short* dst = &Atile[row * 64 + seg];
#if HAVE_ASYNC_LDS
            #pragma unroll
            for (int i = 0; i < 4; ++i)
                g2l_async_b128(src + i * 8, dst + i * 8);
#else
            uint4 tmp[4];
            #pragma unroll
            for (int i = 0; i < 4; ++i) tmp[i] = ((const uint4*)src)[i];
            #pragma unroll
            for (int i = 0; i < 4; ++i) ((uint4*)dst)[i] = tmp[i];
#endif
        }
        {   // B tile: 64x64 bf16, swizzled into B-fragment order
            const int k  = tid >> 2;
            const int n0 = (tid & 3) * 16;
            const uint4* src = (const uint4*)(Bm + (size_t)(k0 + k) * N + nBase + n0);
            union { uint4 q[2]; unsigned short u[16]; } bv;
            bv.q[0] = src[0]; bv.q[1] = src[1];
            unsigned short* bt = &Btile[(k >> 5) * 2048 + (k & 31)];
            #pragma unroll
            for (int i = 0; i < 16; ++i) {
                const int n = n0 + i;
                bt[(n >> 4) * 512 + (n & 15) * 32] = bv.u[i];
            }
        }
        if (k0 + 64 < K) {   // gfx1250 global_prefetch_b8 for the next tiles
            __builtin_prefetch(A + (size_t)(mBase + (tid >> 1)) * K + k0 + 64, 0, 0);
            int pk = k0 + 64 + (tid >> 2); if (pk > K - 1) pk = K - 1;
            __builtin_prefetch(Bm + (size_t)pk * N + nBase, 0, 0);
        }
#if HAVE_ASYNC_LDS
        wait_async0();
#endif
        __syncthreads();

        const int arow = wave * 16 + l15;
        #pragma unroll
        for (int c = 0; c < 2; ++c) {
            Frag af;   // A frag: lane = row, K = hlf*16 + {0..15}
            af.q[0] = *(const uint4*)&Atile[arow * 64 + c * 32 + hlf * 16];
            af.q[1] = *(const uint4*)&Atile[arow * 64 + c * 32 + hlf * 16 + 8];
            #pragma unroll
            for (int nt = 0; nt < 4; ++nt) {
                Frag bf;   // B frag: lane = col, K = hlf*16 + {0..15}
                const uint4* bp =
                    (const uint4*)&Btile[c * 2048 + nt * 512 + l15 * 32 + hlf * 16];
                bf.q[0] = bp[0]; bf.q[1] = bp[1];
                acc[nt] = wmma_bf16(af, bf, acc[nt]);
            }
        }
    }

    // Epilogue. C layout: lane = col (l15), VGPR r = row hlf*8+r.
    #pragma unroll
    for (int nt = 0; nt < 4; ++nt) {
        const int col = nBase + nt * 16 + l15;
        const float bv = HAS_BIAS ? bias[col] : 0.f;
        #pragma unroll
        for (int r = 0; r < 8; ++r) {
            const int row = mBase + wave * 16 + hlf * 8 + r;
            float v = acc[nt][r] + bv;
            if (RELU) v = fmaxf(v, 0.f);
            if (HAS_RES) v += res[(size_t)row * N + col];
            if (STORE_BF16)
                ((unsigned short*)Cout)[(size_t)row * N + col] = f2bf(v);
            else
                ((float*)Cout)[(size_t)row * N + col] = v;
        }
    }
}

// ---------------------------------------------------------------------------
// Flash attention: grid (B*H, T/64), block 128 (4 waves x 16 query rows).
// qkv bf16 [token, 3C] (q | k | v, head-major inside each).  scale = C^-1/2.
// ---------------------------------------------------------------------------
__global__ __launch_bounds__(128)
void attn_kernel(const unsigned short* __restrict__ qkv,
                 unsigned short* __restrict__ att) {
    __shared__ unsigned short VT[64 * 32];       // V^T tile: [dim][key]
    __shared__ unsigned short Pb[4 * 16 * 32];   // per-wave P staging [row][key]
    const int tid  = threadIdx.x;
    const int wave = tid >> 5;
    const int lane = tid & 31;
    const int hlf  = lane >> 4;
    const int l15  = lane & 15;
    const int bh    = blockIdx.x;
    const int b     = bh / NH;
    const int h     = bh % NH;
    const int qtile = blockIdx.y;
    const size_t tokBase = (size_t)b * NT;
    const int qrow0 = qtile * 64 + wave * 16;
    const float scale = 0.03608439182435161f;    // 1/sqrt(768)

    // Q fragments (two 16x32 K-chunks over D=64), loaded once from global.
    Frag qf[2];
    {
        const size_t rowoff = (tokBase + qrow0 + l15) * LDQKV + h * ND;
        #pragma unroll
        for (int dc = 0; dc < 2; ++dc) {
            const uint4* p = (const uint4*)(qkv + rowoff + dc * 32 + hlf * 16);
            qf[dc].q[0] = p[0];      // K = hlf*16 + 0..7
            qf[dc].q[1] = p[1];      // K = hlf*16 + 8..15
        }
    }

    v8f O[4];
    float m_[8], l_[8];
    #pragma unroll
    for (int nt = 0; nt < 4; ++nt)
        #pragma unroll
        for (int r = 0; r < 8; ++r) O[nt][r] = 0.f;
    #pragma unroll
    for (int r = 0; r < 8; ++r) { m_[r] = -1e30f; l_[r] = 0.f; }

    const int nkt = qtile * 2 + 2;               // key tiles of 32 (causal bound)
    for (int kt = 0; kt < nkt; ++kt) {
        __syncthreads();
        {   // stage V^T (32 keys x 64 dims -> [dim][key]), 16 elems/thread
            const int key  = tid >> 2;
            const int dim0 = (tid & 3) * 16;
            const uint4* src = (const uint4*)(qkv + (tokBase + kt * 32 + key) * LDQKV
                                              + 2 * NC + h * ND + dim0);
            union { uint4 q[2]; unsigned short u[16]; } vv;
            vv.q[0] = src[0]; vv.q[1] = src[1];
            #pragma unroll
            for (int i = 0; i < 16; ++i) VT[(dim0 + i) * 32 + key] = vv.u[i];
        }
        __syncthreads();

        // Batch-load all 4 K fragments, then issue the 4 WMMAs.
        // K B-frag: lane = key col, 16 contiguous dims of that key's row.
        Frag kf[2][2];
        #pragma unroll
        for (int kc = 0; kc < 2; ++kc) {
            const size_t krow =
                (tokBase + kt * 32 + kc * 16 + l15) * LDQKV + NC + h * ND;
            #pragma unroll
            for (int dc = 0; dc < 2; ++dc) {
                const uint4* p = (const uint4*)(qkv + krow + dc * 32 + hlf * 16);
                kf[kc][dc].q[0] = p[0]; kf[kc][dc].q[1] = p[1];
            }
        }
        v8f S[2];
        #pragma unroll
        for (int kc = 0; kc < 2; ++kc)
            #pragma unroll
            for (int r = 0; r < 8; ++r) S[kc][r] = 0.f;
        #pragma unroll
        for (int dc = 0; dc < 2; ++dc)
            #pragma unroll
            for (int kc = 0; kc < 2; ++kc)
                S[kc] = wmma_bf16(qf[dc], kf[kc][dc], S[kc]);

        // scale + causal mask + online softmax (per-row state, halves independent)
        const int qrg  = qtile * 64 + wave * 16 + hlf * 8;
        const int col0 = kt * 32 + l15;
        const int col1 = kt * 32 + 16 + l15;
        float pn0[8], pn1[8], alpha[8];
        #pragma unroll
        for (int r = 0; r < 8; ++r) {
            const int qr = qrg + r;
            const float s0 = (col0 <= qr) ? S[0][r] * scale : -1e30f;
            const float s1 = (col1 <= qr) ? S[1][r] * scale : -1e30f;
            float t = fmaxf(s0, s1);
            #pragma unroll
            for (int msk = 1; msk < 16; msk <<= 1)
                t = fmaxf(t, __shfl_xor(t, msk, 32));
            const float mn = fmaxf(m_[r], t);
            alpha[r] = __expf(m_[r] - mn);
            m_[r] = mn;
            pn0[r] = (s0 > -1e29f) ? __expf(s0 - mn) : 0.f;
            pn1[r] = (s1 > -1e29f) ? __expf(s1 - mn) : 0.f;
            float rs = pn0[r] + pn1[r];
            #pragma unroll
            for (int msk = 1; msk < 16; msk <<= 1)
                rs += __shfl_xor(rs, msk, 32);
            l_[r] = l_[r] * alpha[r] + rs;
        }
        #pragma unroll
        for (int nt = 0; nt < 4; ++nt)
            #pragma unroll
            for (int r = 0; r < 8; ++r) O[nt][r] *= alpha[r];

        // P: C-layout -> A-layout through per-wave LDS staging
        unsigned short* Pw = &Pb[wave * 512];
        #pragma unroll
        for (int r = 0; r < 8; ++r) {
            Pw[(hlf * 8 + r) * 32 + l15]      = f2bf(pn0[r]);
            Pw[(hlf * 8 + r) * 32 + 16 + l15] = f2bf(pn1[r]);
        }
        __syncthreads();
        Frag pf;
        {
            const uint4* p = (const uint4*)&Pw[l15 * 32 + hlf * 16];
            pf.q[0] = p[0]; pf.q[1] = p[1];
        }
        // O += P * V   (B-frag: lane = dim col, K = 16 contiguous keys from VT)
        #pragma unroll
        for (int nt = 0; nt < 4; ++nt) {
            Frag vf;
            const uint4* p = (const uint4*)&VT[(nt * 16 + l15) * 32 + hlf * 16];
            vf.q[0] = p[0]; vf.q[1] = p[1];
            O[nt] = wmma_bf16(pf, vf, O[nt]);
        }
    }

    // finalize: divide by l, concat-head store bf16
    #pragma unroll
    for (int r = 0; r < 8; ++r) {
        const float inv = 1.f / l_[r];
        const size_t tokRow = tokBase + qtile * 64 + wave * 16 + hlf * 8 + r;
        #pragma unroll
        for (int nt = 0; nt < 4; ++nt)
            att[tokRow * NC + h * ND + nt * 16 + l15] = f2bf(O[nt][r] * inv);
    }
}

// ---------------------------------------------------------------------------
// Host-side orchestration
// ---------------------------------------------------------------------------
extern "C" void kernel_launch(void* const* d_in, const int* in_sizes, int n_in,
                              void* d_out, int out_size, void* d_ws, size_t ws_size,
                              hipStream_t stream) {
    (void)in_sizes; (void)n_in; (void)out_size; (void)ws_size;
    const float* x    = (const float*)d_in[0];
    const float* Wq   = (const float*)d_in[1];
    const float* Wk   = (const float*)d_in[2];
    const float* Wv   = (const float*)d_in[3];
    const float* Wp   = (const float*)d_in[4];
    const float* bp   = (const float*)d_in[5];
    const float* W1   = (const float*)d_in[6];
    const float* b1   = (const float*)d_in[7];
    const float* W2   = (const float*)d_in[8];
    const float* b2   = (const float*)d_in[9];
    const float* ln1g = (const float*)d_in[10];
    const float* ln1b = (const float*)d_in[11];
    const float* ln2g = (const float*)d_in[12];
    const float* ln2b = (const float*)d_in[13];

    char* w = (char*)d_ws;
    auto alloc = [&](size_t bytes) -> void* {
        void* p = (void*)w;
        w += (bytes + 255) & ~(size_t)255;
        return p;
    };
    unsigned short* Wqkv_bf = (unsigned short*)alloc((size_t)NC * LDQKV * 2);
    unsigned short* Wp_bf   = (unsigned short*)alloc((size_t)NC * NC * 2);
    unsigned short* W1_bf   = (unsigned short*)alloc((size_t)NC * NF * 2);
    unsigned short* W2_bf   = (unsigned short*)alloc((size_t)NF * NC * 2);
    unsigned short* h1_bf   = (unsigned short*)alloc((size_t)NTOK * NC * 2);
    unsigned short* qkv_bf  = (unsigned short*)alloc((size_t)NTOK * LDQKV * 2);
    unsigned short* att_bf  = (unsigned short*)alloc((size_t)NTOK * NC * 2);
    float*          x1      = (float*)alloc((size_t)NTOK * NC * 4);
    unsigned short* h2_bf   = (unsigned short*)alloc((size_t)NTOK * NC * 2);
    unsigned short* ff1_bf  = (unsigned short*)alloc((size_t)NTOK * NF * 2);

    // 1) weight convert/pack to bf16
    {
        int n = NC * LDQKV;
        pack_qkv_kernel<<<(n + 255) / 256, 256, 0, stream>>>(Wq, Wk, Wv, Wqkv_bf);
        n = NC * NC;
        f2bf_kernel<<<(n + 255) / 256, 256, 0, stream>>>(Wp, Wp_bf, n);
        n = NC * NF;
        f2bf_kernel<<<(n + 255) / 256, 256, 0, stream>>>(W1, W1_bf, n);
        f2bf_kernel<<<(n + 255) / 256, 256, 0, stream>>>(W2, W2_bf, n);
    }
    // 2) LN1
    ln_kernel<<<NTOK, 256, 0, stream>>>(x, ln1g, ln1b, h1_bf);
    // 3) QKV GEMM: [4096,768] x [768,2304] -> bf16
    gemm_bf16<true, false, false, false>
        <<<dim3(LDQKV / 64, NTOK / 128), 256, 0, stream>>>(
            h1_bf, Wqkv_bf, qkv_bf, nullptr, nullptr, LDQKV, NC);
    // 4) causal flash attention
    attn_kernel<<<dim3(NB * NH, NT / 64), 128, 0, stream>>>(qkv_bf, att_bf);
    // 5) output projection + bias + residual(x) -> x1 f32
    gemm_bf16<false, false, true, true>
        <<<dim3(NC / 64, NTOK / 128), 256, 0, stream>>>(
            att_bf, Wp_bf, x1, bp, x, NC, NC);
    // 6) LN2
    ln_kernel<<<NTOK, 256, 0, stream>>>(x1, ln2g, ln2b, h2_bf);
    // 7) FF1: relu(h2 @ W1 + b1) -> bf16
    gemm_bf16<true, true, true, false>
        <<<dim3(NF / 64, NTOK / 128), 256, 0, stream>>>(
            h2_bf, W1_bf, ff1_bf, b1, nullptr, NF, NC);
    // 8) FF2: x1 + relu(ff1 @ W2 + b2) -> d_out f32
    gemm_bf16<false, true, true, true>
        <<<dim3(NC / 64, NTOK / 128), 256, 0, stream>>>(
            ff1_bf, W2_bf, (float*)d_out, b2, x1, NC, NF);
}
